// DCGRUCell_30855045055153
// MI455X (gfx1250) — matrix operand order repaired and checked by reading, not compiled, not run on hardware
//
#include <hip/hip_runtime.h>
#include <hip/hip_bf16.h>
#include <math.h>

// ---------------------------------------------------------------------------
// DCGRU cell for MI455X (gfx1250, wave32, WMMA).
// Heavy math: 8x GEMM (4224x2048)·(2048x2048)^T in bf16 WMMA w/ f32 accum.
// Staging: GLOBAL_LOAD_ASYNC_TO_LDS_B128 (ASYNCcnt) when the toolchain has the
// builtin; otherwise classic global_load_b128 -> ds_store_b128 double buffer.
// ---------------------------------------------------------------------------

typedef __attribute__((ext_vector_type(16))) __bf16 v16bf;
typedef __attribute__((ext_vector_type(8)))  float  v8f;
typedef __attribute__((ext_vector_type(4)))  int    v4i;

#define NDIM 2048
#define KDIM 2048
#define MROWS 4224          // B(64) * CIN(66)

#if defined(__has_builtin)
#if __has_builtin(__builtin_amdgcn_global_load_async_to_lds_b128) && \
    __has_builtin(__builtin_amdgcn_s_wait_asynccnt)
#define USE_ASYNC_LDS 1
#endif
#endif
#ifndef USE_ASYNC_LDS
#define USE_ASYNC_LDS 0
#endif

__device__ __forceinline__ unsigned short f2bf(float f) {
  union { float f; unsigned u; } v; v.f = f;
  unsigned r = v.u + 0x7FFFu + ((v.u >> 16) & 1u);   // round-to-nearest-even
  return (unsigned short)(r >> 16);
}
__device__ __forceinline__ float bf2f(unsigned short s) {
  union { unsigned u; float f; } o; o.u = ((unsigned)s) << 16; return o.f;
}
__device__ __forceinline__ v8f vzero8() {
  v8f z;
#pragma unroll
  for (int k = 0; k < 8; ++k) z[k] = 0.0f;
  return z;
}

union Frag16 { v16bf v; uint4 u[2]; };

// A-fragment 16x32 (MxK), bf16, from row-major LDS tile (stride 32 elems).
// Lanes 0-15: row = lane, K 0-7 in v0-3, K 16-23 in v4-7.
// Lanes 16-31: same rows, K 8-15 / K 24-31.  (ISA 7.12.2, 16-bit A 16x32)
__device__ __forceinline__ Frag16 load_a_frag(const unsigned short* tile,
                                              int rowBase, int lane) {
  Frag16 f;
  const unsigned short* p = tile + (rowBase + (lane & 15)) * 32 + ((lane >> 4) << 3);
  f.u[0] = *(const uint4*)(p);
  f.u[1] = *(const uint4*)(p + 16);
  return f;
}
// B-fragment 32x16 (KxN) where B[k,j] = Asup[jglobal, kglobal] -> lane reads a
// pair of 8-element runs (offsets 0 and 16) of Asup row j (tile row-major, stride 32).
// Lanes 0-15: col j = lane, K 0-7 / K 16-23; lanes 16-31: K 8-15 / K 24-31.
__device__ __forceinline__ Frag16 load_b_frag(const unsigned short* tile,
                                              int colBase, int lane) {
  Frag16 f;
  const unsigned short* p = tile + (colBase + (lane & 15)) * 32 + ((lane >> 4) << 3);
  f.u[0] = *(const uint4*)(p);
  f.u[1] = *(const uint4*)(p + 16);
  return f;
}

#if USE_ASYNC_LDS
typedef __attribute__((address_space(1))) v4i* glb_v4i_ptr;
typedef __attribute__((address_space(3))) v4i* lds_v4i_ptr;
__device__ __forceinline__ void async_copy16B(const unsigned short* g,
                                              unsigned short* l) {
  __builtin_amdgcn_global_load_async_to_lds_b128(
      (glb_v4i_ptr)g, (lds_v4i_ptr)l, 0, 0);
}
#endif

// ---------------------------------------------------------------------------
// GEMM: Out[m,j] = alpha * sum_k X[m,k]*A[j,k]  (- Sub[m,j] if Sub != null)
// X: M x 2048 bf16 row-major, A: 2048 x 2048 bf16 row-major, Out bf16.
// Block tile 128x128, 256 threads (8 waves), wave tile 64x32 (4x2 frags),
// double-buffered LDS staging.
// ---------------------------------------------------------------------------
__global__ void __launch_bounds__(256, 1)
dcgru_gemm_bt_bf16(const unsigned short* __restrict__ X,
                   const unsigned short* __restrict__ A,
                   unsigned short* __restrict__ Out,
                   const unsigned short* __restrict__ Sub,
                   float alpha) {
  __shared__ __align__(16) unsigned short sX[2][128 * 32];
  __shared__ __align__(16) unsigned short sA[2][128 * 32];

  const int tid  = threadIdx.x;
  const int lane = tid & 31;
  const int wv   = tid >> 5;
  const int wM   = wv >> 2;                 // 0..1  (64-row slabs)
  const int wN   = wv & 3;                  // 0..3  (32-col slabs)
  const long m0  = (long)blockIdx.y * 128;
  const long n0  = (long)blockIdx.x * 128;

  // cooperative loader geometry: 256 threads x 2 chunks of 16B cover 128x32 bf16
  const int rA = tid >> 2;                  // 0..63
  const int cA = (tid & 3) << 3;            // 0,8,16,24

  const unsigned short* Xp0 = X + (m0 + rA) * (long)KDIM + cA;
  const unsigned short* Xp1 = X + (m0 + rA + 64) * (long)KDIM + cA;
  const unsigned short* Ap0 = A + (n0 + rA) * (long)KDIM + cA;
  const unsigned short* Ap1 = A + (n0 + rA + 64) * (long)KDIM + cA;

  v8f acc[4][2];
#pragma unroll
  for (int i = 0; i < 4; ++i)
#pragma unroll
    for (int j = 0; j < 2; ++j) acc[i][j] = vzero8();

  const int NK = KDIM / 32;                 // 64 k-steps

#if USE_ASYNC_LDS
  // ---- direct global->LDS DMA path (ASYNCcnt) ----
  async_copy16B(Xp0, &sX[0][rA * 32 + cA]);
  async_copy16B(Xp1, &sX[0][(rA + 64) * 32 + cA]);
  async_copy16B(Ap0, &sA[0][rA * 32 + cA]);
  async_copy16B(Ap1, &sA[0][(rA + 64) * 32 + cA]);
  __builtin_amdgcn_s_wait_asynccnt(0);
  __syncthreads();

  for (int t = 0; t < NK; ++t) {
    const int buf = t & 1;
    if (t + 1 < NK) {                       // DMA next tile while we compute
      const int nb = buf ^ 1;
      const long kk = (long)(t + 1) * 32;
      async_copy16B(Xp0 + kk, &sX[nb][rA * 32 + cA]);
      async_copy16B(Xp1 + kk, &sX[nb][(rA + 64) * 32 + cA]);
      async_copy16B(Ap0 + kk, &sA[nb][rA * 32 + cA]);
      async_copy16B(Ap1 + kk, &sA[nb][(rA + 64) * 32 + cA]);
    }

    Frag16 af[4], bfr[2];
#pragma unroll
    for (int i = 0; i < 4; ++i) af[i] = load_a_frag(sX[buf], wM * 64 + i * 16, lane);
#pragma unroll
    for (int j = 0; j < 2; ++j) bfr[j] = load_b_frag(sA[buf], wN * 32 + j * 16, lane);

#pragma unroll
    for (int i = 0; i < 4; ++i)
#pragma unroll
      for (int j = 0; j < 2; ++j)
        acc[i][j] = __builtin_amdgcn_wmma_f32_16x16x32_bf16(
            false, af[i].v, false, bfr[j].v, (short)0, acc[i][j], false, false);

    if (t + 1 < NK) __builtin_amdgcn_s_wait_asynccnt(0);
    __syncthreads();
  }
#else
  // ---- fallback: VGPR-staged double buffer ----
  {
    uint4 x0 = *(const uint4*)(Xp0), x1 = *(const uint4*)(Xp1);
    uint4 a0 = *(const uint4*)(Ap0), a1 = *(const uint4*)(Ap1);
    *(uint4*)&sX[0][rA * 32 + cA]        = x0;
    *(uint4*)&sX[0][(rA + 64) * 32 + cA] = x1;
    *(uint4*)&sA[0][rA * 32 + cA]        = a0;
    *(uint4*)&sA[0][(rA + 64) * 32 + cA] = a1;
  }
  __syncthreads();

  for (int t = 0; t < NK; ++t) {
    const int buf = t & 1;
    uint4 nx0, nx1, na0, na1;
    if (t + 1 < NK) {
      const long kk = (long)(t + 1) * 32;
      nx0 = *(const uint4*)(Xp0 + kk);
      nx1 = *(const uint4*)(Xp1 + kk);
      na0 = *(const uint4*)(Ap0 + kk);
      na1 = *(const uint4*)(Ap1 + kk);
      if (t + 2 < NK) {                     // global_prefetch_b8 for t+2 tiles
        __builtin_prefetch(Xp0 + kk + 32, 0, 1);
        __builtin_prefetch(Ap0 + kk + 32, 0, 1);
      }
    }

    Frag16 af[4], bfr[2];
#pragma unroll
    for (int i = 0; i < 4; ++i) af[i] = load_a_frag(sX[buf], wM * 64 + i * 16, lane);
#pragma unroll
    for (int j = 0; j < 2; ++j) bfr[j] = load_b_frag(sA[buf], wN * 32 + j * 16, lane);

#pragma unroll
    for (int i = 0; i < 4; ++i)
#pragma unroll
      for (int j = 0; j < 2; ++j)
        acc[i][j] = __builtin_amdgcn_wmma_f32_16x16x32_bf16(
            false, af[i].v, false, bfr[j].v, (short)0, acc[i][j], false, false);

    __syncthreads();
    if (t + 1 < NK) {
      const int nb = buf ^ 1;
      *(uint4*)&sX[nb][rA * 32 + cA]        = nx0;
      *(uint4*)&sX[nb][(rA + 64) * 32 + cA] = nx1;
      *(uint4*)&sA[nb][rA * 32 + cA]        = na0;
      *(uint4*)&sA[nb][(rA + 64) * 32 + cA] = na1;
      __syncthreads();
    }
  }
#endif

  // epilogue: C/D layout = lane(0-15): N=lane, M=v;  lane(16-31): M=v+8
  const int cSel = lane & 15;
  const int rSel = (lane >> 4) << 3;
#pragma unroll
  for (int i = 0; i < 4; ++i) {
#pragma unroll
    for (int j = 0; j < 2; ++j) {
      const long mr0 = m0 + wM * 64 + i * 16 + rSel;
      const long nc  = n0 + wN * 32 + j * 16 + cSel;
#pragma unroll
      for (int v = 0; v < 8; ++v) {
        const long idx = (mr0 + v) * (long)NDIM + nc;
        float val = alpha * acc[i][j][v];
        if (Sub) val -= bf2f(Sub[idx]);
        Out[idx] = f2bf(val);
      }
    }
  }
}

// ---------------------------------------------------------------------------
// Pack concat(inputs, hx) -> bf16 feature matrix F0 (4224 x 2048)
// ---------------------------------------------------------------------------
__global__ void dcgru_build_x0(const float* __restrict__ inputs,
                               const float* __restrict__ hx,
                               unsigned short* __restrict__ F0) {
  const int m = blockIdx.y;                       // 0..4223
  const int n = blockIdx.x * 256 + threadIdx.x;   // 0..2047
  const int b = m / 66, c = m % 66;
  const float v = (c < 2) ? inputs[((long)b * 2 + c) * NDIM + n]
                          : hx[((long)b * 64 + (c - 2)) * NDIM + n];
  F0[(long)m * NDIM + n] = f2bf(v);
}

__global__ void dcgru_cvt_bf16(const float* __restrict__ src,
                               unsigned short* __restrict__ dst) {
  const long i = (long)blockIdx.x * 256 + threadIdx.x;
  dst[i] = f2bf(src[i]);
}

// ---------------------------------------------------------------------------
// r/u projection: 330-channel fp32 matvec per (b,n) column + sigmoid.
// Writes G0 = concat(inputs, r*hx) IN PLACE over F block 0 (reads of F
// complete before writes; each thread owns its column) and u -> u_buf.
// NOTE: F/G0 intentionally NOT __restrict__ (they alias).
// ---------------------------------------------------------------------------
__global__ void __launch_bounds__(128, 1)
dcgru_proj_ru(const unsigned short* F,            // 5 mats contiguous
              const float* __restrict__ W_ru,     // 330 x 128
              const float* __restrict__ b_ru,     // 128
              const float* __restrict__ inputs,   // 64x2x2048
              const float* __restrict__ hx,       // 64x64x2048
              unsigned short* G0,                 // aliases F block 0
              float* __restrict__ u_buf) {        // 64x64x2048
  const long g = (long)blockIdx.x * 128 + threadIdx.x;
  const int n = (int)(g & (NDIM - 1));
  const int b = (int)(g >> 11);
  const long MSZ = (long)MROWS * NDIM;

  float acc[128];
#pragma unroll
  for (int o = 0; o < 128; ++o) acc[o] = b_ru[o];

  for (int q = 0; q < 5; ++q) {
    const unsigned short* Fq = F + q * MSZ + ((long)b * 66) * NDIM + n;
    const float* Wq = W_ru + (long)(q * 66) * 128;
    for (int c = 0; c < 66; ++c) {
      const float x = bf2f(Fq[(long)c * NDIM]);
      const float* w = Wq + c * 128;               // uniform -> s_loads
#pragma unroll
      for (int o = 0; o < 128; ++o) acc[o] = fmaf(x, w[o], acc[o]);
    }
  }

  // all F reads done; now write G0 (same memory, same column)
  G0[((long)b * 66 + 0) * NDIM + n] = f2bf(inputs[((long)b * 2 + 0) * NDIM + n]);
  G0[((long)b * 66 + 1) * NDIM + n] = f2bf(inputs[((long)b * 2 + 1) * NDIM + n]);
#pragma unroll 4
  for (int o = 0; o < 64; ++o) {
    const float r = 1.0f / (1.0f + expf(-acc[o]));
    const float u = 1.0f / (1.0f + expf(-acc[64 + o]));
    const float h = hx[((long)b * 64 + o) * NDIM + n];
    G0[((long)b * 66 + 2 + o) * NDIM + n] = f2bf(r * h);
    u_buf[((long)b * 64 + o) * NDIM + n]  = u;
  }
}

// ---------------------------------------------------------------------------
// Candidate projection + GRU combine: out = u*hx + (1-u)*tanh(proj_c)
// ---------------------------------------------------------------------------
__global__ void __launch_bounds__(128, 1)
dcgru_proj_c(const unsigned short* __restrict__ Gf,  // 5 mats contiguous
             const float* __restrict__ W_c,          // 330 x 64
             const float* __restrict__ b_c,          // 64
             const float* __restrict__ u_buf,
             const float* __restrict__ hx,
             float* __restrict__ out) {
  const long g = (long)blockIdx.x * 128 + threadIdx.x;
  const int n = (int)(g & (NDIM - 1));
  const int b = (int)(g >> 11);
  const long MSZ = (long)MROWS * NDIM;

  float acc[64];
#pragma unroll
  for (int o = 0; o < 64; ++o) acc[o] = b_c[o];

  for (int q = 0; q < 5; ++q) {
    const unsigned short* Gq = Gf + q * MSZ + ((long)b * 66) * NDIM + n;
    const float* Wq = W_c + (long)(q * 66) * 64;
    for (int c = 0; c < 66; ++c) {
      const float x = bf2f(Gq[(long)c * NDIM]);
      const float* w = Wq + c * 64;
#pragma unroll
      for (int o = 0; o < 64; ++o) acc[o] = fmaf(x, w[o], acc[o]);
    }
  }

#pragma unroll 4
  for (int o = 0; o < 64; ++o) {
    const float Cv = tanhf(acc[o]);
    const long hidx = ((long)b * 64 + o) * NDIM + n;
    const float u = u_buf[hidx];
    out[hidx] = u * hx[hidx] + (1.0f - u) * Cv;
  }
}

// ---------------------------------------------------------------------------
extern "C" void kernel_launch(void* const* d_in, const int* in_sizes, int n_in,
                              void* d_out, int out_size, void* d_ws, size_t ws_size,
                              hipStream_t stream) {
  const float* inputs   = (const float*)d_in[0];
  const float* hx       = (const float*)d_in[1];
  const float* support0 = (const float*)d_in[2];
  const float* support1 = (const float*)d_in[3];
  const float* W_ru     = (const float*)d_in[4];
  const float* b_ru     = (const float*)d_in[5];
  const float* W_c      = (const float*)d_in[6];
  const float* b_c      = (const float*)d_in[7];
  float* out = (float*)d_out;

  const long MSZ = (long)MROWS * NDIM;               // 8,650,752 elems / feat mat
  unsigned short* F   = (unsigned short*)d_ws;       // 5 * MSZ  (feat mats, contiguous)
  unsigned short* A0b = F + 5 * MSZ;                 // 2048*2048 bf16
  unsigned short* A1b = A0b + (long)NDIM * NDIM;
  float* u_buf = (float*)(A1b + (long)NDIM * NDIM);  // 64*64*2048 fp32
  // total workspace ~137 MB

  dcgru_build_x0<<<dim3(NDIM / 256, MROWS), 256, 0, stream>>>(inputs, hx, F);
  dcgru_cvt_bf16<<<(NDIM * NDIM) / 256, 256, 0, stream>>>(support0, A0b);
  dcgru_cvt_bf16<<<(NDIM * NDIM) / 256, 256, 0, stream>>>(support1, A1b);

  const dim3 gg(NDIM / 128, MROWS / 128);            // 16 x 33 blocks

  // --- diffusion conv #1 over concat(inputs, hx) ---
  dcgru_gemm_bt_bf16<<<gg, 256, 0, stream>>>(F,           A0b, F + 1 * MSZ, nullptr, 1.0f);
  dcgru_gemm_bt_bf16<<<gg, 256, 0, stream>>>(F + 1 * MSZ, A0b, F + 2 * MSZ, F,       2.0f);
  dcgru_gemm_bt_bf16<<<gg, 256, 0, stream>>>(F,           A1b, F + 3 * MSZ, nullptr, 1.0f);
  dcgru_gemm_bt_bf16<<<gg, 256, 0, stream>>>(F + 3 * MSZ, A1b, F + 4 * MSZ, F,       2.0f);

  // r/u gates; rewrites F block 0 as G0 = concat(inputs, r*hx)
  dcgru_proj_ru<<<(64 * NDIM) / 128, 128, 0, stream>>>(F, W_ru, b_ru, inputs, hx, F, u_buf);

  // --- diffusion conv #2 over G0 (reuses the same buffers) ---
  dcgru_gemm_bt_bf16<<<gg, 256, 0, stream>>>(F,           A0b, F + 1 * MSZ, nullptr, 1.0f);
  dcgru_gemm_bt_bf16<<<gg, 256, 0, stream>>>(F + 1 * MSZ, A0b, F + 2 * MSZ, F,       2.0f);
  dcgru_gemm_bt_bf16<<<gg, 256, 0, stream>>>(F,           A1b, F + 3 * MSZ, nullptr, 1.0f);
  dcgru_gemm_bt_bf16<<<gg, 256, 0, stream>>>(F + 3 * MSZ, A1b, F + 4 * MSZ, F,       2.0f);

  // candidate + GRU combine -> d_out
  dcgru_proj_c<<<(64 * NDIM) / 128, 128, 0, stream>>>(F, W_c, b_c, u_buf, hx, out);
}
